// ENCODER_33148557590872
// MI455X (gfx1250) — matrix operand order repaired and checked by reading, not compiled, not run on hardware
//
#include <hip/hip_runtime.h>
#include <math.h>

#define IN_DIM 64
#define H1D    128
#define H2D    256
#define LATD   64

typedef __attribute__((ext_vector_type(16))) __bf16 v16bf;
typedef __attribute__((ext_vector_type(8)))  __bf16 v8bf;
typedef __attribute__((ext_vector_type(4)))  __bf16 v4bf;
typedef __attribute__((ext_vector_type(8)))  float  v8f;

// ---------------- utility kernels ----------------

__global__ __launch_bounds__(256) void k_zero(float* __restrict__ p, int n) {
  int i = blockIdx.x * blockDim.x + threadIdx.x;
  if (i < n) p[i] = 0.0f;
}

__global__ __launch_bounds__(256) void k_deg(const int* __restrict__ dst,
                                             float* __restrict__ deg, int E) {
  int e = blockIdx.x * blockDim.x + threadIdx.x;
  if (e < E) atomicAdd(&deg[dst[e]], 1.0f);
}

__global__ __launch_bounds__(256) void k_dinv(const float* __restrict__ deg,
                                              float* __restrict__ dinv, int N) {
  int i = blockIdx.x * blockDim.x + threadIdx.x;
  if (i < N) dinv[i] = rsqrtf(1.0f + deg[i]);
}

// f32 row-major -> bf16 row-major, 4 elements/thread (vector load+store)
__global__ __launch_bounds__(256) void k_pack_act(const float* __restrict__ in,
                                                  __bf16* __restrict__ out, int n4) {
  int i = blockIdx.x * blockDim.x + threadIdx.x;
  if (i >= n4) return;
  float4 v = ((const float4*)in)[i];
  v4bf o;
  o[0] = (__bf16)v.x; o[1] = (__bf16)v.y; o[2] = (__bf16)v.z; o[3] = (__bf16)v.w;
  *(v4bf*)(out + 4 * (size_t)i) = o;
}

// Pack W[K,D] f32 into per-lane WMMA B fragments (bf16):
//   frag index idx = (n*(K/32) + kk)*2 + hi holds B[kk*32 + hi*16 + i][n], i=0..15
// so the GEMM reads each lane's fragment as one contiguous 32-byte chunk.
__global__ __launch_bounds__(256) void k_pack_w(const float* __restrict__ W,
                                                __bf16* __restrict__ out,
                                                int K, int D) {
  int idx = blockIdx.x * blockDim.x + threadIdx.x;
  int K32 = K >> 5;
  int total = D * K32 * 2;
  if (idx >= total) return;
  int hi = idx & 1;
  int t = idx >> 1;
  int kk = t % K32;
  int n = t / K32;
  const float* src = W + (size_t)(kk * 32 + hi * 16) * D + n;
  v16bf frag;
#pragma unroll
  for (int i = 0; i < 16; ++i) frag[i] = (__bf16)src[(size_t)i * D];
  *(v16bf*)(out + (size_t)idx * 16) = frag;
}

// ---------------- WMMA GEMM + fused GCN self-loop/bias epilogue ----------------
// C = A[M,K](bf16 row-major) @ Bp(K,D fragment-packed bf16); per wave: 16x64 tile.
// Epilogue: xw = C ; agg = C * dinv[row]^2 + bias[col]
// Layouts used here (wave32; A/B share contiguous k order, C/D as in CDNA5 ISA 7.12.2):
//   A 16x32: lane ml=lane&15 row; hi=lane>>4: a[0..15]=K+hi*16.. (matches packed B)
//   B 32x16: lane ml col; lanes 0-15 K=0..15, lanes 16-31 K=16..31 (pre-packed)
//   C/D 16x16 f32: vgpr j -> row j+hi*8, col ml
template <int K, int D>
__global__ __launch_bounds__(256) void k_gemm(
    const __bf16* __restrict__ A, const __bf16* __restrict__ Bp,
    const float* __restrict__ dinv, const float* __restrict__ bias,
    float* __restrict__ xw, float* __restrict__ agg, int totalWaves) {
  constexpr int NT4 = D / 64;
  constexpr int K32 = K / 32;
  int wave = (blockIdx.x * blockDim.x + threadIdx.x) >> 5;
  if (wave >= totalWaves) return;        // wave-uniform: EXEC all-1 at every WMMA
  int lane = threadIdx.x & 31;
  int tm = wave / NT4;
  int tn = wave - tm * NT4;
  int ml = lane & 15;
  int hi = lane >> 4;

  const __bf16* __restrict__ arow = A + (size_t)(tm * 16 + ml) * K + hi * 16;

  v8f acc[4];
  acc[0] = v8f{}; acc[1] = v8f{}; acc[2] = v8f{}; acc[3] = v8f{};

#pragma unroll
  for (int kk = 0; kk < K32; ++kk) {
    v8bf a0 = *(const v8bf*)(arow + kk * 32);
    v8bf a1 = *(const v8bf*)(arow + kk * 32 + 8);
    v16bf a = __builtin_shufflevector(a0, a1, 0, 1, 2, 3, 4, 5, 6, 7,
                                      8, 9, 10, 11, 12, 13, 14, 15);
#pragma unroll
    for (int sub = 0; sub < 4; ++sub) {
      int n = tn * 64 + sub * 16 + ml;
      v16bf b = *(const v16bf*)(Bp + ((size_t)(n * K32 + kk) * 2 + hi) * 16);
      acc[sub] = __builtin_amdgcn_wmma_f32_16x16x32_bf16(
          false, a, false, b, (short)0, acc[sub], false, false);
    }
  }

  // fused epilogue
  float dv2[8];
#pragma unroll
  for (int j = 0; j < 8; ++j) {
    float dv = dinv[tm * 16 + hi * 8 + j];
    dv2[j] = dv * dv;
  }
#pragma unroll
  for (int sub = 0; sub < 4; ++sub) {
    int col = tn * 64 + sub * 16 + ml;
    float bcol = bias[col];
#pragma unroll
    for (int j = 0; j < 8; ++j) {
      size_t off = (size_t)(tm * 16 + hi * 8 + j) * D + col;
      float v = acc[sub][j];
      xw[off] = v;
      agg[off] = v * dv2[j] + bcol;
    }
  }
}

// ---------------- edge aggregation ----------------
// out[dst] += xw[src] * (dinv[src]*dinv[dst]); thread per (edge, float4).
// Consecutive lanes -> consecutive features of an edge: coalesced; xw is L2-resident.
__global__ __launch_bounds__(256) void k_edge_scatter(
    const int* __restrict__ src, const int* __restrict__ dst,
    const float* __restrict__ dinv, const float* __restrict__ xw,
    float* __restrict__ out, int D4, int total) {
  int idx = blockIdx.x * blockDim.x + threadIdx.x;
  if (idx >= total) return;
  int e = idx / D4;
  int f4 = idx - e * D4;
  int s = src[e];
  int d = dst[e];
  float w = dinv[s] * dinv[d];
  float4 v = ((const float4*)xw)[(size_t)s * D4 + f4];
  float* o = out + ((size_t)d * D4 + f4) * 4;
  atomicAdd(o + 0, v.x * w);
  atomicAdd(o + 1, v.y * w);
  atomicAdd(o + 2, v.z * w);
  atomicAdd(o + 3, v.w * w);
}

// ---------------- BatchNorm (training-mode batch stats) ----------------

// blockDim.x == D; each block reduces a 128-row chunk with coalesced reads.
__global__ void k_bn_stats(const float* __restrict__ y, float* __restrict__ gsum,
                           float* __restrict__ gsumsq, int N, int D, int relu_in) {
  int f = threadIdx.x;
  int r0 = blockIdx.x * 128;
  int r1 = min(r0 + 128, N);
  float s = 0.0f, ss = 0.0f;
  for (int r = r0; r < r1; ++r) {
    float v = y[(size_t)r * D + f];
    if (relu_in) v = fmaxf(v, 0.0f);
    s += v;
    ss += v * v;
  }
  atomicAdd(&gsum[f], s);
  atomicAdd(&gsumsq[f], ss);
}

__global__ void k_bn_finalize(const float* __restrict__ gsum,
                              const float* __restrict__ gsumsq,
                              float* __restrict__ mean, float* __restrict__ rstd,
                              int D, float invN) {
  int f = blockIdx.x * blockDim.x + threadIdx.x;
  if (f >= D) return;
  float m = gsum[f] * invN;
  float v = gsumsq[f] * invN - m * m;   // biased variance
  mean[f] = m;
  rstd[f] = rsqrtf(v + 1e-5f);
}

// Normalize (+relu before/after) and emit bf16 activations for the next GEMM.
__global__ __launch_bounds__(256) void k_bn_apply(
    const float* __restrict__ y, const float* __restrict__ mean,
    const float* __restrict__ rstd, const float* __restrict__ g,
    const float* __restrict__ be, __bf16* __restrict__ out, int ND, int D,
    int relu_before, int relu_after) {
  int i = blockIdx.x * blockDim.x + threadIdx.x;
  if (i >= ND) return;
  int f = i % D;
  float v = y[i];
  if (relu_before) v = fmaxf(v, 0.0f);
  v = (v - mean[f]) * rstd[f] * g[f] + be[f];
  if (relu_after) v = fmaxf(v, 0.0f);
  out[i] = (__bf16)v;
}

// ---------------- host orchestration ----------------

static inline int cdiv_i(int a, int b) { return (a + b - 1) / b; }

static void launch_gemm(int K, int D, const __bf16* A, const __bf16* Bp,
                        const float* dinv, const float* bias, float* xw,
                        float* agg, int N, hipStream_t stream) {
  int totalWaves = (N / 16) * (D / 64);
  int blocks = cdiv_i(totalWaves * 32, 256);
  if (K == 64 && D == 128)
    k_gemm<64, 128><<<blocks, 256, 0, stream>>>(A, Bp, dinv, bias, xw, agg, totalWaves);
  else if (K == 128 && D == 256)
    k_gemm<128, 256><<<blocks, 256, 0, stream>>>(A, Bp, dinv, bias, xw, agg, totalWaves);
  else if (K == 256 && D == 64)
    k_gemm<256, 64><<<blocks, 256, 0, stream>>>(A, Bp, dinv, bias, xw, agg, totalWaves);
}

extern "C" void kernel_launch(void* const* d_in, const int* in_sizes, int n_in,
                              void* d_out, int out_size, void* d_ws, size_t ws_size,
                              hipStream_t stream) {
  const int N = in_sizes[0] / IN_DIM;
  const int E = in_sizes[1] / 2;

  const float* x   = (const float*)d_in[0];
  const int*   src = (const int*)d_in[1];
  const int*   dst = src + E;
  const float* mw1 = (const float*)d_in[2];  const float* mb1 = (const float*)d_in[3];
  const float* mg1 = (const float*)d_in[4];  const float* mbe1= (const float*)d_in[5];
  const float* mw2 = (const float*)d_in[6];  const float* mb2 = (const float*)d_in[7];
  const float* mg2 = (const float*)d_in[8];  const float* mbe2= (const float*)d_in[9];
  const float* mw3 = (const float*)d_in[10]; const float* mb3 = (const float*)d_in[11];
  const float* lw1 = (const float*)d_in[12]; const float* lb1 = (const float*)d_in[13];
  const float* lg1 = (const float*)d_in[14]; const float* lbe1= (const float*)d_in[15];
  const float* lw2 = (const float*)d_in[16]; const float* lb2 = (const float*)d_in[17];
  const float* lg2 = (const float*)d_in[18]; const float* lbe2= (const float*)d_in[19];
  const float* lw3 = (const float*)d_in[20]; const float* lb3 = (const float*)d_in[21];

  // ---- workspace carve-up (f32 regions first, then bf16; all 32B-aligned) ----
  float* f     = (float*)d_ws;
  float* deg   = f;
  float* dinv  = f + N;
  float* xw    = f + 2 * (size_t)N;                  // [N,256] f32
  float* agg   = xw + (size_t)N * H2D;               // [N,256] f32
  float* gsum  = agg + (size_t)N * H2D;              // [256]
  float* gsumsq= gsum + H2D;
  float* meanb = gsumsq + H2D;
  float* rstdb = meanb + H2D;
  __bf16* xbf  = (__bf16*)(rstdb + H2D);             // [N,64]  bf16 (packed x)
  __bf16* hbf  = xbf + (size_t)N * IN_DIM;           // [N,256] bf16 (activations)
  __bf16* wp   = hbf + (size_t)N * H2D;              // [<=128*256] bf16 (packed W)

  // degrees + dinv (shared by both branches)
  k_zero<<<cdiv_i(N, 256), 256, 0, stream>>>(deg, N);
  k_deg<<<cdiv_i(E, 256), 256, 0, stream>>>(dst, deg, E);
  k_dinv<<<cdiv_i(N, 256), 256, 0, stream>>>(deg, dinv, N);

  // pack x once (used by both branches)
  k_pack_act<<<cdiv_i(N * IN_DIM / 4, 256), 256, 0, stream>>>(x, xbf, N * IN_DIM / 4);

  auto conv = [&](const __bf16* featbf, int K, const float* W, const float* b,
                  int D, float* aggOut) {
    int packTotal = D * (K / 32) * 2;
    k_pack_w<<<cdiv_i(packTotal, 256), 256, 0, stream>>>(W, wp, K, D);
    launch_gemm(K, D, featbf, wp, dinv, b, xw, aggOut, N, stream);
    int D4 = D / 4;
    int total = E * D4;
    k_edge_scatter<<<cdiv_i(total, 256), 256, 0, stream>>>(src, dst, dinv, xw,
                                                           aggOut, D4, total);
  };

  auto bn = [&](const float* y, const float* g, const float* be, __bf16* out,
                int D, int relu_before, int relu_after) {
    k_zero<<<1, 512, 0, stream>>>(gsum, 2 * H2D);    // gsum & gsumsq contiguous
    k_bn_stats<<<cdiv_i(N, 128), D, 0, stream>>>(y, gsum, gsumsq, N, D, relu_before);
    k_bn_finalize<<<1, D, 0, stream>>>(gsum, gsumsq, meanb, rstdb, D, 1.0f / (float)N);
    k_bn_apply<<<cdiv_i(N * D, 256), 256, 0, stream>>>(y, meanb, rstdb, g, be, out,
                                                       N * D, D, relu_before, relu_after);
  };

  float* mu = (float*)d_out;
  float* lg = mu + (size_t)N * LATD;

  // encoder_mu: conv -> BN -> ReLU (x2), conv
  conv(xbf, IN_DIM, mw1, mb1, H1D, agg);
  bn(agg, mg1, mbe1, hbf, H1D, /*relu_before=*/0, /*relu_after=*/1);
  conv(hbf, H1D, mw2, mb2, H2D, agg);
  bn(agg, mg2, mbe2, hbf, H2D, 0, 1);
  conv(hbf, H2D, mw3, mb3, LATD, mu);

  // encoder_log: conv -> ReLU -> BN (x2), conv
  conv(xbf, IN_DIM, lw1, lb1, H1D, agg);
  bn(agg, lg1, lbe1, hbf, H1D, /*relu_before=*/1, /*relu_after=*/0);
  conv(hbf, H1D, lw2, lb2, H2D, agg);
  bn(agg, lg2, lbe2, hbf, H2D, 1, 0);
  conv(hbf, H2D, lw3, lb3, LATD, lg);
}
